// ScaledDotProductAttention_72722386256060
// MI455X (gfx1250) — compile-verified, matrix operand order checked by reading, not run on hardware
//
#include <hip/hip_runtime.h>

// ---------------------------------------------------------------------------
// Scaled dot-product attention for MI455X (gfx1250), wave32 + WMMA bf16.
//  B=8 H=16 S=1024 DK=64.  Outputs: context | attn | attn_score (flat f32).
// ---------------------------------------------------------------------------

#define ATTN_B  8
#define ATTN_H  16
#define ATTN_S  1024
#define ATTN_DK 64
#define NEG_INF (-1.0e9f)

typedef __attribute__((ext_vector_type(16))) __bf16 v16bf;
typedef __attribute__((ext_vector_type(8)))  float  v8f;

static __device__ __forceinline__ v8f wmma_bf16(v16bf a, v16bf b, v8f c) {
    // D = A(16x32 bf16) * B(32x16 bf16) + C(16x16 f32)
    return __builtin_amdgcn_wmma_f32_16x16x32_bf16(
        /*neg_a=*/false, a, /*neg_b=*/false, b,
        /*c_mod=*/(short)0, c, /*reuse_a=*/false, /*reuse_b=*/false);
}

// Build a 16-element bf16 fragment from the two contiguous 32B runs that the
// CDNA5 16-bit A/B layout assigns to one lane:
//   elements 0..7  <- dk = base + 0..7      (vectors a,b)
//   elements 8..15 <- dk = base + 16 + 0..7 (vectors c,d)
static __device__ __forceinline__ v16bf make_frag(float4 a, float4 b,
                                                  float4 c, float4 d) {
    v16bf f;
    f[0]  = (__bf16)a.x; f[1]  = (__bf16)a.y; f[2]  = (__bf16)a.z; f[3]  = (__bf16)a.w;
    f[4]  = (__bf16)b.x; f[5]  = (__bf16)b.y; f[6]  = (__bf16)b.z; f[7]  = (__bf16)b.w;
    f[8]  = (__bf16)c.x; f[9]  = (__bf16)c.y; f[10] = (__bf16)c.z; f[11] = (__bf16)c.w;
    f[12] = (__bf16)d.x; f[13] = (__bf16)d.y; f[14] = (__bf16)d.z; f[15] = (__bf16)d.w;
    return f;
}

__global__ void sdpa_zero_kernel(float* __restrict__ p, int n) {
    int i = blockIdx.x * blockDim.x + threadIdx.x;
    if (i < n) p[i] = 0.0f;
}

__global__ void sdpa_fused_kernel(const float* __restrict__ Q,
                                  const float* __restrict__ K,
                                  const float* __restrict__ V,
                                  const unsigned char* __restrict__ mask,
                                  float* __restrict__ ctx_out,
                                  float* __restrict__ attn_out,
                                  float* __restrict__ score_out) {
    const int tid  = threadIdx.x;       // 0..127 (4 wave32s)
    const int lane = tid & 31;
    const int wave = tid >> 5;          // 0..3
    const int g    = lane >> 4;         // lane group (0 or 1)
    const int nl   = lane & 15;         // row (A) / column (B,C,D) index
    const int bh   = blockIdx.y;        // 0..B*H-1
    const int q0   = blockIdx.x * 16;   // query tile base

    extern __shared__ float smem[];
    float* sScores = smem;                   // [16][1024]  64 KB
    float* sQ      = smem + 16 * 1024;       // [16][64]     4 KB
    float* sRed    = sQ + 16 * 64;           // [16][8]
    float* sRowMax = sRed + 128;             // [16]
    float* sRowInv = sRowMax + 16;           // [16]

    // ---- stage Q tile (coalesced, b128) --------------------------------
    {
        const float4* Qsrc = (const float4*)(Q + ((size_t)bh * ATTN_S + q0) * ATTN_DK);
        float4* q4 = (float4*)sQ;
        for (int i = tid; i < 16 * ATTN_DK / 4; i += 128) q4[i] = Qsrc[i];
    }
    __syncthreads();

    // ---- build A fragments of Q (each wave keeps its own copy) ---------
    // Lane's runs in a 64-wide row, as float4 indices: base g*2, plus +4
    // (dk+16); second 32-chunk is +8 float4s.
    v16bf aQ0, aQ1;
    {
        const float4* qr = (const float4*)(sQ + nl * ATTN_DK);
        const int b0 = g * 2;
        aQ0 = make_frag(qr[b0],     qr[b0 + 1],  qr[b0 + 4],  qr[b0 + 5]);
        aQ1 = make_frag(qr[b0 + 8], qr[b0 + 9],  qr[b0 + 12], qr[b0 + 13]);
    }

    // ---- S = (Q K^T) * scale, masked; scores -> LDS --------------------
    const float scale = 0.125f;  // 1/sqrt(64)
    for (int kt = wave; kt < ATTN_S / 16; kt += 4) {
        const int k0 = kt * 16;
        const float4* Kr =
            (const float4*)(K + ((size_t)bh * ATTN_S + k0 + nl) * ATTN_DK);
        const int b0 = g * 2;
        const v16bf bK0 = make_frag(Kr[b0],     Kr[b0 + 1], Kr[b0 + 4],  Kr[b0 + 5]);
        const v16bf bK1 = make_frag(Kr[b0 + 8], Kr[b0 + 9], Kr[b0 + 12], Kr[b0 + 13]);

        // Prefetch next k-tile's mask bytes (streaming read, NT-ish).
        if (kt + 4 < ATTN_S / 16) {
            __builtin_prefetch(
                &mask[((size_t)(bh * ATTN_S + q0 + 8 * g)) * ATTN_S + k0 + 64 + nl],
                0, 1);
        }

        v8f c = {};
        c = wmma_bf16(aQ0, bK0, c);
        c = wmma_bf16(aQ1, bK1, c);
#pragma unroll
        for (int i = 0; i < 8; ++i) {
            const int M = i + 8 * g;
            float sc = c[i] * scale;
            const unsigned char mv =
                mask[((size_t)(bh * ATTN_S + q0 + M)) * ATTN_S + k0 + nl];
            if (mv) sc = NEG_INF;
            sScores[M * ATTN_S + k0 + nl] = sc;
        }
    }
    __syncthreads();

    // ---- softmax over each of the 16 rows (8 threads per row) ----------
    const int r   = tid >> 3;   // row 0..15
    const int sub = tid & 7;    // 8 workers per row
    {
        float m = -3.4e38f;
        for (int i = sub; i < ATTN_S; i += 8)
            m = fmaxf(m, sScores[r * ATTN_S + i]);
        sRed[r * 8 + sub] = m;
    }
    __syncthreads();
    if (sub == 0) {
        float m = sRed[r * 8];
#pragma unroll
        for (int j = 1; j < 8; ++j) m = fmaxf(m, sRed[r * 8 + j]);
        sRowMax[r] = m;
    }
    __syncthreads();
    {
        const float rm = sRowMax[r];
        float s = 0.0f;
        for (int i = sub; i < ATTN_S; i += 8) {
            const float p = __expf(sScores[r * ATTN_S + i] - rm);
            sScores[r * ATTN_S + i] = p;
            s += p;
        }
        sRed[r * 8 + sub] = s;
    }
    __syncthreads();
    if (sub == 0) {
        float s = sRed[r * 8];
#pragma unroll
        for (int j = 1; j < 8; ++j) s += sRed[r * 8 + j];
        sRowInv[r] = 1.0f / s;
    }
    __syncthreads();
    {
        const float inv = sRowInv[r];
        for (int i = sub; i < ATTN_S; i += 8)
            sScores[r * ATTN_S + i] *= inv;
    }
    __syncthreads();

    // ---- stream attn tile to HBM (b128, coalesced; flat since S=1024) --
    {
        const float4* s4 = (const float4*)sScores;
        float4* a4 = (float4*)(attn_out + ((size_t)(bh * ATTN_S + q0)) * ATTN_S);
        for (int i = tid; i < 16 * ATTN_S / 4; i += 128) a4[i] = s4[i];
    }

    // ---- attn_score: column sums over the 16 local rows ----------------
    // Hardware f32 atomic add (global_atomic_add_f32), relaxed/agent scope.
#pragma unroll
    for (int cc = 0; cc < 2; ++cc) {
        const int c4 = tid * 2 + cc;            // float4 column group 0..255
        float4 s = make_float4(0.f, 0.f, 0.f, 0.f);
        const float4* s4 = (const float4*)sScores;
#pragma unroll
        for (int rr = 0; rr < 16; ++rr) {
            const float4 v = s4[rr * (ATTN_S / 4) + c4];
            s.x += v.x; s.y += v.y; s.z += v.z; s.w += v.w;
        }
        float* dst = score_out + (size_t)bh * ATTN_S + c4 * 4;
        __hip_atomic_fetch_add(dst + 0, s.x, __ATOMIC_RELAXED, __HIP_MEMORY_SCOPE_AGENT);
        __hip_atomic_fetch_add(dst + 1, s.y, __ATOMIC_RELAXED, __HIP_MEMORY_SCOPE_AGENT);
        __hip_atomic_fetch_add(dst + 2, s.z, __ATOMIC_RELAXED, __HIP_MEMORY_SCOPE_AGENT);
        __hip_atomic_fetch_add(dst + 3, s.w, __ATOMIC_RELAXED, __HIP_MEMORY_SCOPE_AGENT);
    }

    // ---- context = P @ V : wave w owns DK columns [16w, 16w+16) --------
    {
        const int n0 = wave * 16;
        const float4* pr = (const float4*)(sScores + nl * ATTN_S);
        v8f acc = {};
        for (int ch = 0; ch < ATTN_S / 32; ++ch) {
            const int kb = ch * 32;
            const int p0 = (kb + g * 8) / 4;    // float4 index of first run
            const v16bf aP = make_frag(pr[p0], pr[p0 + 1], pr[p0 + 4], pr[p0 + 5]);
            v16bf bV;
#pragma unroll
            for (int e = 0; e < 16; ++e) {
                const int key = kb + ((e & 8) ? 16 : 0) + g * 8 + (e & 7);
                bV[e] = (__bf16)V[((size_t)bh * ATTN_S + key) * ATTN_DK + n0 + nl];
            }
            acc = wmma_bf16(aP, bV, acc);
        }
#pragma unroll
        for (int i = 0; i < 8; ++i) {
            const int M = i + 8 * g;
            ctx_out[((size_t)(bh * ATTN_S + q0 + M)) * ATTN_DK + n0 + nl] = acc[i];
        }
    }
}

extern "C" void kernel_launch(void* const* d_in, const int* in_sizes, int n_in,
                              void* d_out, int out_size, void* d_ws, size_t ws_size,
                              hipStream_t stream) {
    (void)in_sizes; (void)n_in; (void)out_size; (void)d_ws; (void)ws_size;

    const float* Q = (const float*)d_in[0];
    const float* K = (const float*)d_in[1];
    const float* V = (const float*)d_in[2];
    const unsigned char* mask = (const unsigned char*)d_in[3];

    float* ctx   = (float*)d_out;                                        // B*H*S*DK
    float* attn  = ctx  + (size_t)ATTN_B * ATTN_H * ATTN_S * ATTN_DK;    // B*H*S*S
    float* score = attn + (size_t)ATTN_B * ATTN_H * ATTN_S * ATTN_S;     // B*H*S

    // attn_score is accumulated atomically across q-tiles: zero it first.
    const int nScore = ATTN_B * ATTN_H * ATTN_S;   // 131072
    sdpa_zero_kernel<<<(nScore + 255) / 256, 256, 0, stream>>>(score, nScore);

    const size_t smemBytes =
        (size_t)(16 * 1024 + 16 * 64 + 128 + 16 + 16) * sizeof(float);  // 70272 B
    hipFuncSetAttribute(reinterpret_cast<const void*>(sdpa_fused_kernel),
                        hipFuncAttributeMaxDynamicSharedMemorySize,
                        (int)smemBytes);

    dim3 grid(ATTN_S / 16, ATTN_B * ATTN_H);   // 64 x 128 blocks
    dim3 block(128);                           // 4 wave32s
    sdpa_fused_kernel<<<grid, block, smemBytes, stream>>>(
        Q, K, V, mask, ctx, attn, score);
}
